// KNNModelAE_56581899157603
// MI455X (gfx1250) — compile-verified, matrix operand order checked by reading, not run on hardware
//
#include <hip/hip_runtime.h>
#include <hip/hip_bf16.h>

typedef __attribute__((ext_vector_type(16))) __bf16       v16bf;
typedef __attribute__((ext_vector_type(8)))  float        v8f;
typedef __attribute__((ext_vector_type(4)))  unsigned int u32x4;

#define B_      8
#define E_      768
#define HL      32
#define WL      32
#define NQ      (B_*HL*WL)        // 8192 queries
#define NB      20000             // bank rows
#define OUTW    512
#define NSLICE  5
#define TILES   (NB/16)           // 1250
#define TPS     (TILES/NSLICE)    // 250 tiles per slice
#define KC      (E_/32)           // 24 K-chunks of 32

#define QPW     32                // queries per wave (two 16-wide B tiles)
#define WAVES   4                 // waves per workgroup
#define QPB     (QPW*WAVES)       // 128 queries per workgroup
#define LROW    776               // LDS row stride in ushorts (768 + 8 pad -> conflict-free)

#define FLOATMAX 3.402823466e+38f

// ---------- helpers ----------
__device__ __forceinline__ unsigned short f2bf(float f) {
    unsigned int u = __float_as_uint(f);
    u += 0x7fffu + ((u >> 16) & 1u);            // round-to-nearest-even
    return (unsigned short)(u >> 16);
}
__device__ __forceinline__ float bf2f(unsigned short h) {
    return __uint_as_float(((unsigned int)h) << 16);
}
__device__ __forceinline__ void insert3(float& m0, float& m1, float& m2, float v) {
    float a0 = fminf(v,  m0);
    float h0 = fmaxf(v,  m0);
    float a1 = fminf(h0, m1);
    float h1 = fmaxf(h0, m1);
    float a2 = fminf(h1, m2);
    m0 = a0; m1 = a1; m2 = a2;
}

union V16 { v16bf bf; u32x4 q[2]; };
union F8  { v8f v; float f[8]; };

// ---------- stage 1: bank -> bf16 hi/lo + row norms ----------
__global__ void prep_bank(const float* __restrict__ src,
                          unsigned short* __restrict__ hi,
                          unsigned short* __restrict__ lo,
                          float* __restrict__ nrm) {
    __shared__ float red[256];
    const int row = blockIdx.x;
    const float* s = src + (size_t)row * E_;
    float acc = 0.f;
    for (int e = threadIdx.x; e < E_; e += 256) {
        float x = s[e];
        unsigned short h = f2bf(x);
        unsigned short l = f2bf(x - bf2f(h));
        hi[(size_t)row * E_ + e] = h;
        lo[(size_t)row * E_ + e] = l;
        acc += x * x;
    }
    red[threadIdx.x] = acc;
    __syncthreads();
    for (int s2 = 128; s2 > 0; s2 >>= 1) {
        if (threadIdx.x < (unsigned)s2) red[threadIdx.x] += red[threadIdx.x + s2];
        __syncthreads();
    }
    if (threadIdx.x == 0) nrm[row] = red[0];
}

// ---------- stage 2: embeddings [B,E,H,W] -> queries [NQ,E] bf16 hi/lo + norms ----------
__global__ void prep_query(const float* __restrict__ emb,
                           unsigned short* __restrict__ hi,
                           unsigned short* __restrict__ lo,
                           float* __restrict__ nrm) {
    __shared__ float red[256];
    const int qi = blockIdx.x;               // b*1024 + y*32 + x
    const int b  = qi >> 10;
    const int yx = qi & 1023;
    const float* s = emb + (size_t)b * E_ * (HL * WL) + yx;
    float acc = 0.f;
    for (int e = threadIdx.x; e < E_; e += 256) {
        float x = s[(size_t)e * (HL * WL)];
        unsigned short h = f2bf(x);
        unsigned short l = f2bf(x - bf2f(h));
        hi[(size_t)qi * E_ + e] = h;
        lo[(size_t)qi * E_ + e] = l;
        acc += x * x;
    }
    red[threadIdx.x] = acc;
    __syncthreads();
    for (int s2 = 128; s2 > 0; s2 >>= 1) {
        if (threadIdx.x < (unsigned)s2) red[threadIdx.x] += red[threadIdx.x + s2];
        __syncthreads();
    }
    if (threadIdx.x == 0) nrm[qi] = red[0];
}

// ---------- stage 3: LDS-staged WMMA distance GEMM + streaming top-3 ----------
// grid = (NQ/QPB, NSLICE), block = 128 (4 waves).
// Each 16-row bank tile (bf16 hi+lo) is staged once into LDS and consumed by all
// 4 waves; each wave owns 32 queries (two WMMA B tiles, 2 f32 accumulators).
// D[m=bank row][n=query]; query n lives in lanes {n, n+16} across the 8 acc VGPRs.
__global__ void __launch_bounds__(128)
knn_wmma(const unsigned short* __restrict__ qh, const unsigned short* __restrict__ ql,
         const unsigned short* __restrict__ bh, const unsigned short* __restrict__ bl,
         const float* __restrict__ bn2, float* __restrict__ partials) {
    __shared__ unsigned short ldsH[16 * LROW];
    __shared__ unsigned short ldsL[16 * LROW];

    const int tid   = threadIdx.x;
    const int lane  = tid & 31;
    const int wave  = tid >> 5;
    const int slice = blockIdx.y;
    const int qbase = blockIdx.x * QPB + wave * QPW;
    const int r16   = lane & 15;
    const int khalf = (lane >> 4) << 3;      // 0 or 8 (ushort offset into K-chunk halves)

    const unsigned short* q0h = qh + (size_t)(qbase      + r16) * E_ + khalf;
    const unsigned short* q0l = ql + (size_t)(qbase      + r16) * E_ + khalf;
    const unsigned short* q1h = qh + (size_t)(qbase + 16 + r16) * E_ + khalf;
    const unsigned short* q1l = ql + (size_t)(qbase + 16 + r16) * E_ + khalf;

    const unsigned short* aH = ldsH + r16 * LROW + khalf;
    const unsigned short* aL = ldsL + r16 * LROW + khalf;

    float a0 = FLOATMAX, a1 = FLOATMAX, a2 = FLOATMAX;   // top-3 for queries qbase+0..15
    float c0 = FLOATMAX, c1 = FLOATMAX, c2 = FLOATMAX;   // top-3 for queries qbase+16..31

    for (int t = 0; t < TPS; ++t) {
        const int j0 = (slice * TPS + t) * 16;

        __syncthreads();                                 // previous tile fully consumed
        // cooperative stage: 16 rows x 1536B (hi) + (lo), b128 granules, padded rows
        #pragma unroll
        for (int i = 0; i < (16 * 96) / 128; ++i) {      // 12 chunks per thread
            const int c   = tid + i * 128;
            const int row = c / 96;
            const int col = (c % 96) * 8;                // ushort offset within row
            *(u32x4*)(ldsH + row * LROW + col) = *(const u32x4*)(bh + (size_t)(j0 + row) * E_ + col);
            *(u32x4*)(ldsL + row * LROW + col) = *(const u32x4*)(bl + (size_t)(j0 + row) * E_ + col);
        }
        __builtin_prefetch(bh + (size_t)(j0 + 16) * E_ + tid * 16, 0, 1);  // next tile
        __syncthreads();

        v8f acc0 = {}; v8f acc1 = {};
        #pragma unroll 2
        for (int kc = 0; kc < KC; ++kc) {
            const int kb = kc * 32;
            V16 Ah, Al, B0h, B0l, B1h, B1l;
            Ah.q[0]  = *(const u32x4*)(aH  + kb);  Ah.q[1]  = *(const u32x4*)(aH  + kb + 16);
            Al.q[0]  = *(const u32x4*)(aL  + kb);  Al.q[1]  = *(const u32x4*)(aL  + kb + 16);
            B0h.q[0] = *(const u32x4*)(q0h + kb);  B0h.q[1] = *(const u32x4*)(q0h + kb + 16);
            B0l.q[0] = *(const u32x4*)(q0l + kb);  B0l.q[1] = *(const u32x4*)(q0l + kb + 16);
            B1h.q[0] = *(const u32x4*)(q1h + kb);  B1h.q[1] = *(const u32x4*)(q1h + kb + 16);
            B1l.q[0] = *(const u32x4*)(q1l + kb);  B1l.q[1] = *(const u32x4*)(q1l + kb + 16);
            // hi*hi + hi*lo + lo*hi (drop lo*lo): ~fp32-accurate dot in bf16 WMMA
            acc0 = __builtin_amdgcn_wmma_f32_16x16x32_bf16(false, Ah.bf, false, B0h.bf, (short)0, acc0, false, false);
            acc0 = __builtin_amdgcn_wmma_f32_16x16x32_bf16(false, Ah.bf, false, B0l.bf, (short)0, acc0, false, false);
            acc0 = __builtin_amdgcn_wmma_f32_16x16x32_bf16(false, Al.bf, false, B0h.bf, (short)0, acc0, false, false);
            acc1 = __builtin_amdgcn_wmma_f32_16x16x32_bf16(false, Ah.bf, false, B1h.bf, (short)0, acc1, false, false);
            acc1 = __builtin_amdgcn_wmma_f32_16x16x32_bf16(false, Ah.bf, false, B1l.bf, (short)0, acc1, false, false);
            acc1 = __builtin_amdgcn_wmma_f32_16x16x32_bf16(false, Al.bf, false, B1h.bf, (short)0, acc1, false, false);
        }
        F8 f0; f0.v = acc0;
        F8 f1; f1.v = acc1;
        const float* p0 = bn2 + j0 + khalf;              // ||b||^2 for rows m = r + khalf
        #pragma unroll
        for (int r = 0; r < 8; ++r) {
            const float bb = p0[r];
            insert3(a0, a1, a2, bb - 2.0f * f0.f[r]);
            insert3(c0, c1, c2, bb - 2.0f * f1.f[r]);
        }
    }

    // merge the two half-wave partial top-3 lists for each query
    {
        float o0 = __shfl_xor(a0, 16, 32);
        float o1 = __shfl_xor(a1, 16, 32);
        float o2 = __shfl_xor(a2, 16, 32);
        insert3(a0, a1, a2, o0); insert3(a0, a1, a2, o1); insert3(a0, a1, a2, o2);
        o0 = __shfl_xor(c0, 16, 32);
        o1 = __shfl_xor(c1, 16, 32);
        o2 = __shfl_xor(c2, 16, 32);
        insert3(c0, c1, c2, o0); insert3(c0, c1, c2, o1); insert3(c0, c1, c2, o2);
    }
    if (lane < 16) {
        float* d0 = partials + ((size_t)(qbase      + lane) * NSLICE + slice) * 3;
        float* d1 = partials + ((size_t)(qbase + 16 + lane) * NSLICE + slice) * 3;
        d0[0] = a0; d0[1] = a1; d0[2] = a2;
        d1[0] = c0; d1[1] = c1; d1[2] = c2;
    }
}

// ---------- stage 4: merge slices, add ||q||^2, sqrt, mean-of-3 ----------
__global__ void merge_topk(const float* __restrict__ partials,
                           const float* __restrict__ q2,
                           float* __restrict__ scores) {
    const int qi = blockIdx.x * blockDim.x + threadIdx.x;
    if (qi >= NQ) return;
    float m0 = FLOATMAX, m1 = FLOATMAX, m2 = FLOATMAX;
    const float* p = partials + (size_t)qi * (NSLICE * 3);
    #pragma unroll
    for (int i = 0; i < NSLICE * 3; ++i) insert3(m0, m1, m2, p[i]);
    const float qq = q2[qi];
    float d0 = sqrtf(fmaxf(m0 + qq, 1e-12f));
    float d1 = sqrtf(fmaxf(m1 + qq, 1e-12f));
    float d2 = sqrtf(fmaxf(m2 + qq, 1e-12f));
    scores[qi] = (d0 + d1 + d2) * (1.0f / 3.0f);
}

// ---------- stage 5: half-pixel bilinear upsample 32x32 -> 512x512 ----------
__global__ void upsample(const float* __restrict__ scores, float* __restrict__ out) {
    const int idx = blockIdx.x * blockDim.x + threadIdx.x;
    if (idx >= B_ * OUTW * OUTW) return;
    const int ox = idx % OUTW;
    const int oy = (idx / OUTW) % OUTW;
    const int b  = idx / (OUTW * OUTW);
    const float sx = (ox + 0.5f) * (32.0f / 512.0f) - 0.5f;
    const float sy = (oy + 0.5f) * (32.0f / 512.0f) - 0.5f;
    int x0 = (int)floorf(sx); float fx = sx - (float)x0;
    int y0 = (int)floorf(sy); float fy = sy - (float)y0;
    int x1 = min(max(x0 + 1, 0), 31); x0 = min(max(x0, 0), 31);
    int y1 = min(max(y0 + 1, 0), 31); y0 = min(max(y0, 0), 31);
    const float* sp = scores + (size_t)b * 1024;
    float v00 = sp[y0 * 32 + x0], v01 = sp[y0 * 32 + x1];
    float v10 = sp[y1 * 32 + x0], v11 = sp[y1 * 32 + x1];
    out[idx] = (1.f - fy) * ((1.f - fx) * v00 + fx * v01)
             +        fy  * ((1.f - fx) * v10 + fx * v11);
}

extern "C" void kernel_launch(void* const* d_in, const int* in_sizes, int n_in,
                              void* d_out, int out_size, void* d_ws, size_t ws_size,
                              hipStream_t stream) {
    (void)in_sizes; (void)n_in; (void)out_size; (void)ws_size;
    const float* emb  = (const float*)d_in[0];   // [8,768,32,32] fp32
    const float* bank = (const float*)d_in[1];   // [20000,768]   fp32
    float* out = (float*)d_out;                  // [8,1,512,512] fp32

    char* ws = (char*)d_ws;
    size_t off = 0;
    auto carve = [&](size_t bytes) -> void* {
        void* p = ws + off;
        off = (off + bytes + 255) & ~(size_t)255;
        return p;
    };
    unsigned short* bh  = (unsigned short*)carve((size_t)NB * E_ * 2);
    unsigned short* bl  = (unsigned short*)carve((size_t)NB * E_ * 2);
    unsigned short* qhp = (unsigned short*)carve((size_t)NQ * E_ * 2);
    unsigned short* qlp = (unsigned short*)carve((size_t)NQ * E_ * 2);
    float* b2       = (float*)carve((size_t)NB * 4);
    float* q2       = (float*)carve((size_t)NQ * 4);
    float* partials = (float*)carve((size_t)NQ * NSLICE * 3 * 4);
    float* scores   = (float*)carve((size_t)NQ * 4);

    prep_bank <<<NB, 256, 0, stream>>>(bank, bh, bl, b2);
    prep_query<<<NQ, 256, 0, stream>>>(emb, qhp, qlp, q2);
    knn_wmma  <<<dim3(NQ / QPB, NSLICE), WAVES * 32, 0, stream>>>(qhp, qlp, bh, bl, b2, partials);
    merge_topk<<<(NQ + 255) / 256, 256, 0, stream>>>(partials, q2, scores);
    upsample  <<<(B_ * OUTW * OUTW + 255) / 256, 256, 0, stream>>>(scores, out);
}
